// Block_21174188770119
// MI455X (gfx1250) — matrix operand order repaired and checked by reading, not compiled, not run on hardware
//
#include <hip/hip_runtime.h>

typedef __attribute__((ext_vector_type(16))) __bf16 v16bf;
typedef __attribute__((ext_vector_type(8)))  float  v8f;
typedef __attribute__((ext_vector_type(4)))  int    v4i_t;

union BF16x16 { v16bf v; uint4 u[2]; };

#define WMMA_BF16(a, b, c) \
  __builtin_amdgcn_wmma_f32_16x16x32_bf16(false, (a), false, (b), (short)0, (c), false, false)

#ifndef __has_builtin
#define __has_builtin(x) 0
#endif
#if __has_builtin(__builtin_amdgcn_global_load_async_to_lds_b128)
#define HAVE_ASYNC_LDS 1
#else
#define HAVE_ASYNC_LDS 0
#endif

#if HAVE_ASYNC_LDS
typedef __attribute__((address_space(1))) v4i_t glob_v4i;   // global (prints as __device__)
typedef __attribute__((address_space(3))) v4i_t lds_v4i;    // LDS

__device__ __forceinline__ void async_copy_b128(const void* g, void* l)
{
  // generic->AS(1): same 64-bit address; generic->AS(3): LDS offset is low 32 bits
  glob_v4i* gp = (glob_v4i*)(uintptr_t)g;
  lds_v4i*  lp = (lds_v4i*)(uint32_t)(uintptr_t)l;
  __builtin_amdgcn_global_load_async_to_lds_b128(gp, lp, 0, 0);
}
#define ASYNC_WAIT0() asm volatile("s_wait_asynccnt 0x0" ::: "memory")
#endif

static constexpr int CN    = 256;
static constexpr int NIMG  = 8;
static constexpr int PIX   = 4096;            // 64*64
static constexpr int MROWS = NIMG * PIX;      // 32768

// ---------------------------------------------------------------------------
// Weight prep: pack f32 weights to bf16 GEMM-friendly layouts
// Wqkv [1536][256] = rows: v_in_w(768) then h_in_w(768)
// Wout [256][512]  = Wout[j][k] = k<256 ? v_out_w[j][k] : h_out_w[j][k-256]
// ---------------------------------------------------------------------------
__global__ void prep_kernel(const float* __restrict__ vw, const float* __restrict__ hw,
                            const float* __restrict__ vow, const float* __restrict__ how,
                            const float* __restrict__ vb, const float* __restrict__ hb,
                            const float* __restrict__ vob, const float* __restrict__ hob,
                            __bf16* __restrict__ Wqkv, __bf16* __restrict__ Wout,
                            float* __restrict__ bqkv, float* __restrict__ bout)
{
  const int idx = blockIdx.x * 256 + threadIdx.x;
  if (idx < 1536 * 256) {
    const float w = (idx < 768 * 256) ? vw[idx] : hw[idx - 768 * 256];
    Wqkv[idx] = (__bf16)w;
  }
  if (idx < 256 * 512) {
    const int j = idx >> 9, k = idx & 511;
    const float w = (k < 256) ? vow[j * 256 + k] : how[j * 256 + (k - 256)];
    Wout[idx] = (__bf16)w;
  }
  if (idx < 1536) bqkv[idx] = (idx < 768) ? vb[idx] : hb[idx - 768];
  if (idx < 256)  bout[idx] = vob[idx] + hob[idx];
}

// ---------------------------------------------------------------------------
// Block reduction (256 threads = 8 waves of 32)
// ---------------------------------------------------------------------------
__device__ inline void block_reduce2(float& s, float& s2, float* sbuf)
{
  #pragma unroll
  for (int m = 16; m >= 1; m >>= 1) {
    s  += __shfl_xor(s,  m, 32);
    s2 += __shfl_xor(s2, m, 32);
  }
  const int wave = threadIdx.x >> 5;
  if ((threadIdx.x & 31) == 0) { sbuf[wave] = s; sbuf[8 + wave] = s2; }
  __syncthreads();
  float ts = 0.f, ts2 = 0.f;
  #pragma unroll
  for (int i = 0; i < 8; ++i) { ts += sbuf[i]; ts2 += sbuf[8 + i]; }
  s = ts; s2 = ts2;
  __syncthreads();
}

// ---------------------------------------------------------------------------
// InstanceNorm #1: per (n,c) over 4096 pixels.
// Writes f32 xn [N,C,H,W] (residual path) and bf16 xp [N*H*W, C] (GEMM A).
// ---------------------------------------------------------------------------
__global__ void instnorm_in_kernel(const float* __restrict__ img,
                                   float* __restrict__ xn,
                                   __bf16* __restrict__ xp)
{
  __shared__ float sbuf[16];
  const int nc = blockIdx.x;
  const int n = nc >> 8, c = nc & 255;
  const float* src = img + (size_t)nc * PIX;
  float vals[16];
  float s = 0.f, s2 = 0.f;
  #pragma unroll
  for (int i = 0; i < 16; ++i) {
    const float x = src[threadIdx.x + i * 256];
    vals[i] = x; s += x; s2 += x * x;
  }
  block_reduce2(s, s2, sbuf);
  const float mu  = s * (1.0f / 4096.0f);
  const float var = s2 * (1.0f / 4096.0f) - mu * mu;
  const float ri  = rsqrtf(var + 1e-5f);
  float*  xo  = xn + (size_t)nc * PIX;
  __bf16* xpo = xp + (size_t)n * PIX * CN + c;
  #pragma unroll
  for (int i = 0; i < 16; ++i) {
    const int p = threadIdx.x + i * 256;
    const float z = (vals[i] - mu) * ri;
    xo[p] = z;
    xpo[(size_t)p * CN] = (__bf16)z;
  }
}

// ---------------------------------------------------------------------------
// bf16 WMMA GEMM: Out[M,N] = A[M,K] @ W[N,K]^T + bias[N]
// Block: 256 threads (8 waves). Tile: 128(M) x 64(N). K step 32.
// Double-buffered B tile staged via GLOBAL_LOAD_ASYNC_TO_LDS_B128 (ASYNCcnt);
// per K-step: s_wait_asynccnt 0 -> barrier -> kick next DMA -> WMMA on current.
// Fragment layouts follow CDNA5 ISA 7.12.2 (16-bit A 16x32, B 32x16).
// ---------------------------------------------------------------------------
template<bool OUT_BF16>
__global__ void gemm_bf16_kernel(const __bf16* __restrict__ A,
                                 const __bf16* __restrict__ W,
                                 const float* __restrict__ bias,
                                 void* __restrict__ Out,
                                 int M, int N, int K)
{
  __shared__ __attribute__((aligned(16))) __bf16 Bs[2][64][32];
  const int tid  = threadIdx.x;
  const int wave = tid >> 5;
  const int lane = tid & 31;
  const int np = lane & 15, g = lane >> 4;
  const int Mbase = blockIdx.x * 128 + wave * 16;
  const int Nbase = blockIdx.y * 64;

  const v8f vz = {0.f, 0.f, 0.f, 0.f, 0.f, 0.f, 0.f, 0.f};
  v8f acc[4] = {vz, vz, vz, vz};

  const int sn = tid >> 2;            // 0..63   B-stage row
  const int sk = (tid & 3) * 8;       // 0,8,16,24
  const __bf16* wsrc = W + (size_t)(Nbase + sn) * K + sk;
  const __bf16* arow0 = A + (size_t)(Mbase + np) * K;

  const int nk = K / 32;
#if HAVE_ASYNC_LDS
  async_copy_b128(wsrc, &Bs[0][sn][sk]);
#endif
  for (int i = 0; i < nk; ++i) {
    const int cur = i & 1;
    const int kb = i * 32;
#if HAVE_ASYNC_LDS
    ASYNC_WAIT0();                 // this wave's B-tile DMA complete
    __syncthreads();               // all waves' DMAs done, old readers retired
    if (i + 1 < nk)                // prefetch next tile into other buffer
      async_copy_b128(wsrc + (i + 1) * 32, &Bs[cur ^ 1][sn][sk]);
#else
    __syncthreads();
    *(uint4*)&Bs[cur][sn][sk] = *(const uint4*)(wsrc + kb);
    __syncthreads();
#endif
    // A fragment: row = Mbase+np; group g: vgprs0-3 k=kb+g*8.., vgprs4-7 k=kb+16+g*8..
    BF16x16 af;
    const __bf16* arow = arow0 + kb;
    __builtin_prefetch(arow + 64, 0, 0);   // global_prefetch_b8 for the A stream
    af.u[0] = *(const uint4*)(arow + g * 8);
    af.u[1] = *(const uint4*)(arow + 16 + g * 8);
    #pragma unroll
    for (int t = 0; t < 4; ++t) {
      BF16x16 bf_;
      const __bf16* bp = &Bs[cur][t * 16 + np][0];
      bf_.u[0] = *(const uint4*)(bp + g * 8); bf_.u[1] = *(const uint4*)(bp + 16 + g * 8);
      acc[t] = WMMA_BF16(af.v, bf_.v, acc[t]);
    }
  }

  // epilogue: C layout -> lane holds N = t*16+np, rows r + 8*g
  #pragma unroll
  for (int t = 0; t < 4; ++t) {
    const int n = Nbase + t * 16 + np;
    const float bv = bias[n];
    #pragma unroll
    for (int r = 0; r < 8; ++r) {
      const int m = Mbase + r + 8 * g;
      const float val = acc[t][r] + bv;
      if (OUT_BF16) ((__bf16*)Out)[(size_t)m * N + n] = (__bf16)val;
      else          ((float*)Out)[(size_t)m * N + n] = val;
    }
  }
}

// ---------------------------------------------------------------------------
// Factored attention core. grid=(512, 2): x = batch-row (n*64 + h|w), y = dir.
// 8 waves per block, one head per wave. Seq len 64, dh 32, scale 1/sqrt(32).
// qkv: bf16 [32768][1536], cols: dir*768 + {q:0, k:256, v:512} + head*32.
// ctx: bf16 [32768][512],  cols: dir*256 + head*32 + d.
// ---------------------------------------------------------------------------
__global__ void attention_kernel(const __bf16* __restrict__ qkv,
                                 __bf16* __restrict__ ctx)
{
  __shared__ __attribute__((aligned(16))) __bf16 Vt[8][32][64];  // per-wave V^T
  __shared__ __attribute__((aligned(16))) __bf16 Pm[8][16][64];  // per-wave probs
  const int b    = blockIdx.x;
  const int dir  = blockIdx.y;
  const int wave = threadIdx.x >> 5;   // head
  const int lane = threadIdx.x & 31;
  const int np = lane & 15, g = lane >> 4;
  const int nimg = b >> 6, rc = b & 63;
  const size_t mbase   = (dir == 0) ? (size_t)b * 64 : ((size_t)nimg * PIX + rc);
  const size_t mstride = (dir == 0) ? 1 : 64;
  const int qcol = dir * 768 + wave * 32;
  const float sc = 0.17677669529663688f;   // 1/sqrt(32)
  const v8f vz = {0.f, 0.f, 0.f, 0.f, 0.f, 0.f, 0.f, 0.f};

  // K fragments (B for S=QK^T): lane col = key seq, vgprs hold dh pairs
  BF16x16 kf[4];
  #pragma unroll
  for (int t = 0; t < 4; ++t) {
    const __bf16* kp = qkv + (mbase + (size_t)(t * 16 + np) * mstride) * 1536 + qcol + 256;
    kf[t].u[0] = *(const uint4*)(kp + g * 8);
    kf[t].u[1] = *(const uint4*)(kp + 16 + g * 8);
  }

  // V transposed into LDS: Vt[d][s] = V[s][d]
  #pragma unroll
  for (int rr = 0; rr < 2; ++rr) {
    const int s = lane * 2 + rr;
    const uint4* vp = (const uint4*)(qkv + (mbase + (size_t)s * mstride) * 1536 + qcol + 512);
    union { uint4 u[4]; __bf16 h[32]; } vv;
    vv.u[0] = vp[0]; vv.u[1] = vp[1]; vv.u[2] = vp[2]; vv.u[3] = vp[3];
    #pragma unroll
    for (int d = 0; d < 32; ++d) Vt[wave][d][s] = vv.h[d];
  }
  __syncthreads();

  for (int mc = 0; mc < 4; ++mc) {
    __syncthreads();   // Pm reuse fence
    // Q A-fragment for rows mc*16 .. mc*16+15
    BF16x16 qf;
    const __bf16* qp = qkv + (mbase + (size_t)(mc * 16 + np) * mstride) * 1536 + qcol;
    qf.u[0] = *(const uint4*)(qp + g * 8);
    qf.u[1] = *(const uint4*)(qp + 16 + g * 8);

    v8f st[4];
    #pragma unroll
    for (int t = 0; t < 4; ++t) st[t] = WMMA_BF16(qf.v, kf[t].v, vz);

    // softmax: each row lives in 16 lanes of one half-wave across 4 tiles
    #pragma unroll
    for (int r = 0; r < 8; ++r) {
      float a0 = st[0][r] * sc, a1 = st[1][r] * sc, a2 = st[2][r] * sc, a3 = st[3][r] * sc;
      float mx = fmaxf(fmaxf(a0, a1), fmaxf(a2, a3));
      #pragma unroll
      for (int xm = 1; xm < 16; xm <<= 1) mx = fmaxf(mx, __shfl_xor(mx, xm, 32));
      a0 = __expf(a0 - mx); a1 = __expf(a1 - mx); a2 = __expf(a2 - mx); a3 = __expf(a3 - mx);
      float sm = a0 + a1 + a2 + a3;
      #pragma unroll
      for (int xm = 1; xm < 16; xm <<= 1) sm += __shfl_xor(sm, xm, 32);
      const float riv = 1.0f / sm;
      const int row = r + 8 * g;
      Pm[wave][row][ 0 + np] = (__bf16)(a0 * riv);
      Pm[wave][row][16 + np] = (__bf16)(a1 * riv);
      Pm[wave][row][32 + np] = (__bf16)(a2 * riv);
      Pm[wave][row][48 + np] = (__bf16)(a3 * riv);
    }
    __syncthreads();

    // ctx = A @ V : K=64 (2 steps of 32), N=32 (2 tiles)
    v8f c0 = vz, c1 = vz;
    #pragma unroll
    for (int s = 0; s < 2; ++s) {
      BF16x16 af, b0, b1;
      const __bf16* pr = &Pm[wave][np][s * 32];
      af.u[0] = *(const uint4*)(pr + g * 8);
      af.u[1] = *(const uint4*)(pr + 16 + g * 8);
      const __bf16* v0p = &Vt[wave][np     ][s * 32];
      const __bf16* v1p = &Vt[wave][np + 16][s * 32];
      b0.u[0] = *(const uint4*)(v0p + g * 8); b0.u[1] = *(const uint4*)(v0p + 16 + g * 8);
      b1.u[0] = *(const uint4*)(v1p + g * 8); b1.u[1] = *(const uint4*)(v1p + 16 + g * 8);
      c0 = WMMA_BF16(af.v, b0.v, c0);
      c1 = WMMA_BF16(af.v, b1.v, c1);
    }

    const int ocol = dir * 256 + wave * 32 + np;
    #pragma unroll
    for (int r = 0; r < 8; ++r) {
      const size_t row = mbase + (size_t)(mc * 16 + r + 8 * g) * mstride;
      ctx[row * 512 + ocol     ] = (__bf16)c0[r];
      ctx[row * 512 + ocol + 16] = (__bf16)c1[r];
    }
  }
}

// ---------------------------------------------------------------------------
// Residual + InstanceNorm #2 + grouped 1x1 MLP (1->4->1 per channel) + residual
// ---------------------------------------------------------------------------
__global__ void finalize_kernel(const float* __restrict__ attn,   // [N*4096][256]
                                const float* __restrict__ xn,     // [N,C,4096]
                                const float* __restrict__ w1, const float* __restrict__ b1,
                                const float* __restrict__ w2, const float* __restrict__ b2,
                                float* __restrict__ out)
{
  __shared__ float sbuf[16];
  const int nc = blockIdx.x;
  const int n = nc >> 8, c = nc & 255;
  const float* xnp = xn + (size_t)nc * PIX;
  const float* ap  = attn + (size_t)n * PIX * CN + c;
  float vals[16];
  float s = 0.f, s2 = 0.f;
  #pragma unroll
  for (int i = 0; i < 16; ++i) {
    const int p = threadIdx.x + i * 256;
    const float t = ap[(size_t)p * CN] + xnp[p];
    vals[i] = t; s += t; s2 += t * t;
  }
  block_reduce2(s, s2, sbuf);
  const float mu  = s * (1.0f / 4096.0f);
  const float var = s2 * (1.0f / 4096.0f) - mu * mu;
  const float ri  = rsqrtf(var + 1e-5f);
  float w1v[4], b1v[4], w2v[4];
  #pragma unroll
  for (int j = 0; j < 4; ++j) {
    w1v[j] = w1[c * 4 + j]; b1v[j] = b1[c * 4 + j]; w2v[j] = w2[c * 4 + j];
  }
  const float b2v = b2[c];
  float* op = out + (size_t)nc * PIX;
  #pragma unroll
  for (int i = 0; i < 16; ++i) {
    const int p = threadIdx.x + i * 256;
    const float z = (vals[i] - mu) * ri;
    float acc = z + b2v;
    #pragma unroll
    for (int j = 0; j < 4; ++j) acc += w2v[j] * fmaxf(z * w1v[j] + b1v[j], 0.f);
    op[p] = acc;
  }
}

// ---------------------------------------------------------------------------
extern "C" void kernel_launch(void* const* d_in, const int* in_sizes, int n_in,
                              void* d_out, int out_size, void* d_ws, size_t ws_size,
                              hipStream_t stream)
{
  const float* images  = (const float*)d_in[0];
  const float* v_in_w  = (const float*)d_in[1];
  const float* v_in_b  = (const float*)d_in[2];
  const float* v_out_w = (const float*)d_in[3];
  const float* v_out_b = (const float*)d_in[4];
  const float* h_in_w  = (const float*)d_in[5];
  const float* h_in_b  = (const float*)d_in[6];
  const float* h_out_w = (const float*)d_in[7];
  const float* h_out_b = (const float*)d_in[8];
  const float* w1 = (const float*)d_in[9];
  const float* b1 = (const float*)d_in[10];
  const float* w2 = (const float*)d_in[11];
  const float* b2 = (const float*)d_in[12];
  float* out = (float*)d_out;

  char* ws = (char*)d_ws;
  size_t off = 0;
  auto alloc = [&](size_t bytes) -> void* {
    void* p = ws + off;
    off = (off + bytes + 255) & ~(size_t)255;
    return p;
  };
  float*  xn   = (float*)  alloc((size_t)NIMG * CN * PIX * 4);   // 33.5 MB
  __bf16* xp   = (__bf16*) alloc((size_t)MROWS * CN * 2);        // 16.8 MB
  __bf16* qkv  = (__bf16*) alloc((size_t)MROWS * 1536 * 2);      // 100.7 MB
  __bf16* ctxb = (__bf16*) alloc((size_t)MROWS * 512 * 2);       // 33.6 MB
  float*  attn = (float*)  alloc((size_t)MROWS * CN * 4);        // 33.5 MB
  __bf16* Wqkv = (__bf16*) alloc(1536ull * 256 * 2);
  __bf16* Wout = (__bf16*) alloc(256ull * 512 * 2);
  float*  bqkv = (float*)  alloc(1536 * 4);
  float*  bout = (float*)  alloc(256 * 4);
  (void)ws_size; (void)in_sizes; (void)n_in; (void)out_size;

  prep_kernel<<<1536, 256, 0, stream>>>(v_in_w, h_in_w, v_out_w, h_out_w,
                                        v_in_b, h_in_b, v_out_b, h_out_b,
                                        Wqkv, Wout, bqkv, bout);
  instnorm_in_kernel<<<NIMG * CN, 256, 0, stream>>>(images, xn, xp);
  gemm_bf16_kernel<true><<<dim3(MROWS / 128, 1536 / 64), 256, 0, stream>>>(
      xp, Wqkv, bqkv, (void*)qkv, MROWS, 1536, 256);
  attention_kernel<<<dim3(512, 2), 256, 0, stream>>>(qkv, ctxb);
  gemm_bf16_kernel<false><<<dim3(MROWS / 128, 256 / 64), 256, 0, stream>>>(
      ctxb, Wout, bout, (void*)attn, MROWS, 256, 512);
  finalize_kernel<<<NIMG * CN, 256, 0, stream>>>(attn, xn, w1, b1, w2, b2, out);
}
